// FFNUpdate_85693187490097
// MI455X (gfx1250) — matrix operand order repaired and checked by reading, not compiled, not run on hardware
//
#include <hip/hip_runtime.h>

#define NN 100000
#define NE 600000
#define DD 128

typedef __attribute__((ext_vector_type(16))) __bf16 v16bf;
typedef __attribute__((ext_vector_type(8)))  float  v8f;

union Frag { v16bf v; unsigned int u[8]; };

__device__ __forceinline__ unsigned short f2bf(float f) {
  unsigned int u = __builtin_bit_cast(unsigned int, f);
  u += 0x7FFFu + ((u >> 16) & 1u);           // round-to-nearest-even
  return (unsigned short)(u >> 16);
}

// reduce each of a[0..7] over the 16-lane half-wave (xor masks 1,2,4,8)
__device__ __forceinline__ void hred(float* a) {
#pragma unroll
  for (int m = 1; m <= 8; m <<= 1)
#pragma unroll
    for (int r = 0; r < 8; ++r) a[r] += __shfl_xor(a[r], m, 32);
}

// ---- single 16-row tile GEMM: C(16x16) += A(16 x 32*kTiles bf16 LDS) @ B(packed) ----
__device__ __forceinline__ v8f mm_lds(const unsigned short* Ab, int stride, int lane,
                                      int kTiles, const unsigned short* Bpk, int n_tile) {
  const int half8 = (lane >= 16) ? 8 : 0;
  const unsigned short* rowp = Ab + (lane & 15) * stride;
  const uint4* Bp = (const uint4*)Bpk + (size_t)(n_tile * kTiles) * 64 + (lane << 1);
  v8f acc = {};
  for (int kt = 0; kt < kTiles; ++kt) {
    Frag a, b;
#pragma unroll
    for (int v = 0; v < 8; ++v) {
      int k = kt * 32 + ((v >> 2) << 4) + half8 + ((v & 3) << 1);
      a.u[v] = *(const unsigned int*)(rowp + k);
    }
    uint4 q0 = Bp[0], q1 = Bp[1];
    b.u[0] = q0.x; b.u[1] = q0.y; b.u[2] = q0.z; b.u[3] = q0.w;
    b.u[4] = q1.x; b.u[5] = q1.y; b.u[6] = q1.z; b.u[7] = q1.w;
    __builtin_prefetch(Bp + 64, 0, 3);
    acc = __builtin_amdgcn_wmma_f32_16x16x32_bf16(false, a.v, false, b.v,
                                                  (short)0, acc, false, false);
    Bp += 64;
  }
  return acc;
}

// ---- dual 16-row tiles (rows 0-15 and 16-31) sharing each B fragment ----
__device__ __forceinline__ void mm_lds2(const unsigned short* Ab, int stride, int lane,
                                        int kTiles, const unsigned short* Bpk, int n_tile,
                                        v8f& acc0, v8f& acc1) {
  const int half8 = (lane >= 16) ? 8 : 0;
  const unsigned short* rowp0 = Ab + (lane & 15) * stride;
  const unsigned short* rowp1 = rowp0 + 16 * stride;
  const uint4* Bp = (const uint4*)Bpk + (size_t)(n_tile * kTiles) * 64 + (lane << 1);
  v8f z = {};
  acc0 = z; acc1 = z;
  for (int kt = 0; kt < kTiles; ++kt) {
    Frag a0, a1, b;
#pragma unroll
    for (int v = 0; v < 8; ++v) {
      int k = kt * 32 + ((v >> 2) << 4) + half8 + ((v & 3) << 1);
      a0.u[v] = *(const unsigned int*)(rowp0 + k);
      a1.u[v] = *(const unsigned int*)(rowp1 + k);
    }
    uint4 q0 = Bp[0], q1 = Bp[1];
    b.u[0] = q0.x; b.u[1] = q0.y; b.u[2] = q0.z; b.u[3] = q0.w;
    b.u[4] = q1.x; b.u[5] = q1.y; b.u[6] = q1.z; b.u[7] = q1.w;
    __builtin_prefetch(Bp + 64, 0, 3);
    acc0 = __builtin_amdgcn_wmma_f32_16x16x32_bf16(false, a0.v, false, b.v,
                                                   (short)0, acc0, false, false);
    acc1 = __builtin_amdgcn_wmma_f32_16x16x32_bf16(false, a1.v, false, b.v,
                                                   (short)0, acc1, false, false);
    Bp += 64;
  }
}

__global__ __launch_bounds__(256) void zero_kernel(float* __restrict__ p, int n) {
  int i = (blockIdx.x * 256 + threadIdx.x) * 4;
  if (i < n) *(float4*)(p + i) = make_float4(0.f, 0.f, 0.f, 0.f);
}

__global__ __launch_bounds__(256) void h2bf_kernel(const float* __restrict__ H,
                                                   unsigned short* __restrict__ Hbf, int n) {
  int i = (blockIdx.x * 256 + threadIdx.x) * 4;
  if (i < n) {
    float4 f4 = *(const float4*)(H + i);
    ushort4 u = make_ushort4(f2bf(f4.x), f2bf(f4.y), f2bf(f4.z), f2bf(f4.w));
    *(ushort4*)(Hbf + i) = u;
  }
}

// fp32 row-major W[K][N] -> bf16 WMMA-B fragment stream; frag f = n_tile*kTiles + k_tile
__global__ __launch_bounds__(256) void pack_b_kernel(const float* __restrict__ W,
                                                     unsigned short* __restrict__ out,
                                                     int K, int N) {
  int kTiles = K >> 5, nTiles = N >> 4;
  int frags = kTiles * nTiles;
  int f = blockIdx.x * 8 + (threadIdx.x >> 5);
  if (f >= frags) return;
  int lane = threadIdx.x & 31;
  int n_t = f / kTiles, k_t = f - n_t * kTiles;
  int n  = (n_t << 4) + (lane & 15);
  int kb = (k_t << 5) + ((lane >= 16) ? 8 : 0);
  unsigned int* op = (unsigned int*)out + (size_t)f * 256 + lane * 8;
#pragma unroll
  for (int v = 0; v < 8; ++v) {
    int k = kb + ((v >> 2) << 4) + ((v & 3) << 1);
    unsigned int lo = f2bf(W[(size_t)k * N + n]);
    unsigned int hi = f2bf(W[(size_t)(k + 1) * N + n]);
    op[v] = lo | (hi << 16);
  }
}

// ---------------- edge stage: messages + scatter-add ----------------
__global__ __launch_bounds__(256) void edge_kernel(
    const unsigned short* __restrict__ Hbf, const long long* __restrict__ ei,
    const unsigned short* __restrict__ epPk, const float* __restrict__ ep_b,
    const float* __restrict__ ep_g, const float* __restrict__ ep_bb,
    float* __restrict__ agg) {
  __shared__ __align__(16) unsigned short Abuf[16 * 264];   // 16 x 256 bf16, +8 pad
  __shared__ int   tgtS[16];
  __shared__ float part[8][16], partq[8][16];
  __shared__ float meanS[16], rstdS[16];

  const int tid = threadIdx.x, wave = tid >> 5, lane = tid & 31;
  const int nloc = lane & 15, half8 = (lane >= 16) ? 8 : 0;
  const int eBase = blockIdx.x * 16;

  {   // async gather of concat(Hbf[src], Hbf[tgt]) -> LDS (2 x 16B per thread)
    int r = tid >> 4;
    long long e = eBase + r;
    long long sidx = ei[e];
    long long tidx = ei[(long long)NE + e];
    if ((tid & 15) == 0) tgtS[r] = (int)tidx;
    unsigned long long gs = (unsigned long long)(Hbf + (size_t)sidx * 128 + (tid & 15) * 8);
    unsigned long long gt = (unsigned long long)(Hbf + (size_t)tidx * 128 + (tid & 15) * 8);
    unsigned lds1 = (unsigned)(unsigned long long)(&Abuf[r * 264 + (tid & 15) * 8]);
    unsigned lds2 = lds1 + 256;
    asm volatile("global_load_async_to_lds_b128 %0, %1, off"
                 :: "v"(lds1), "v"(gs) : "memory");
    asm volatile("global_load_async_to_lds_b128 %0, %1, off"
                 :: "v"(lds2), "v"(gt) : "memory");
  }
  asm volatile("s_wait_asynccnt 0" ::: "memory");
  __syncthreads();

  v8f acc = mm_lds(Abuf, 264, lane, 8, epPk, wave);  // [16 x 16] slice, K=256
  const int n = wave * 16 + nloc;
  float bias = ep_b[n];
  float vals[8], rs[8], rss[8];
#pragma unroll
  for (int r = 0; r < 8; ++r) {
    float x = acc[r] + bias; x = x > 0.f ? x : 0.f;
    vals[r] = x; rs[r] = x; rss[r] = x * x;
  }
  hred(rs); hred(rss);
  if (nloc == 0)
#pragma unroll
    for (int r = 0; r < 8; ++r) { part[wave][half8 + r] = rs[r]; partq[wave][half8 + r] = rss[r]; }
  __syncthreads();
  if (tid < 16) {
    float s = 0.f, sq = 0.f;
    for (int w = 0; w < 8; ++w) { s += part[w][tid]; sq += partq[w][tid]; }
    float mean = s * (1.f / 128.f);
    float var  = sq * (1.f / 128.f) - mean * mean;
    meanS[tid] = mean; rstdS[tid] = rsqrtf(var + 1e-5f);
  }
  __syncthreads();
  float gg = ep_g[n], bb = ep_bb[n];
#pragma unroll
  for (int r = 0; r < 8; ++r) {
    int m = half8 + r;
    float y = (vals[r] - meanS[m]) * rstdS[m] * gg + bb;
    atomicAdd(&agg[(size_t)tgtS[m] * 128 + n], y);
  }
}

// ---------------- node stage: gate + gated FFN, 32 nodes/block ----------------
__global__ __launch_bounds__(256) void node_kernel(
    const float* __restrict__ H, const unsigned short* __restrict__ Hbf,
    const float* __restrict__ agg,
    const unsigned short* __restrict__ gatePk, const float* __restrict__ gate_b,
    const float* __restrict__ gate_g, const float* __restrict__ gate_bb,
    const float* __restrict__ ln1_g, const float* __restrict__ ln1_b,
    const unsigned short* __restrict__ lin1Pk, const float* __restrict__ lin1_b,
    const unsigned short* __restrict__ sw1Pk, const unsigned short* __restrict__ sw2Pk,
    const unsigned short* __restrict__ sw3Pk,
    const unsigned short* __restrict__ lin2Pk, const float* __restrict__ lin2_b,
    const unsigned short* __restrict__ lin3Pk, const float* __restrict__ lin3_b,
    const float* __restrict__ ln2_g, const float* __restrict__ ln2_b,
    float* __restrict__ out) {
  extern __shared__ __align__(16) char smem[];
  unsigned short* Abuf = (unsigned short*)(smem);              // 32 x 520 bf16  (33280 B)
  unsigned short* Bbuf = (unsigned short*)(smem + 33280);      // 32 x 520 bf16
  float* Hf    = (float*)(smem + 66560);                       // 32 x 128 f32
  float* Gf    = (float*)(smem + 82944);                       // 32 x 128 f32
  float* part  = (float*)(smem + 99328);                       // 8 x 32
  float* partq = (float*)(smem + 100352);                      // 8 x 32
  float* meanS = (float*)(smem + 101376);                      // 32
  float* rstdS = (float*)(smem + 101504);                      // 32
  float* hsum  = (float*)(smem + 101632);                      // 32
  float* hsq   = (float*)(smem + 101760);                      // 32

  const int tid = threadIdx.x, wave = tid >> 5, lane = tid & 31;
  const int nloc = lane & 15, half8 = (lane >= 16) ? 8 : 0;
  const int nodeBase = blockIdx.x * 32;
  const int n = wave * 16 + nloc;

  {   // stage 1: 8 threads/row, 32 cols each
    int r = tid >> 3, c0 = (tid & 7) * 32;
    size_t row = (size_t)(nodeBase + r) * 128;
    if (c0 < 128) {
#pragma unroll
      for (int i = 0; i < 32; i += 4)
        *(float4*)(Hf + r * 128 + c0 + i) = *(const float4*)(H + row + c0 + i);
      unsigned ldsb = (unsigned)(unsigned long long)(&Abuf[r * 520 + c0]);
      unsigned long long g = (unsigned long long)(Hbf + row + c0);
#pragma unroll
      for (int i = 0; i < 4; ++i)
        asm volatile("global_load_async_to_lds_b128 %0, %1, off"
                     :: "v"(ldsb + i * 16), "v"(g + (unsigned long long)(i * 16)) : "memory");
    } else {
      int c1 = c0 - 128;
#pragma unroll
      for (int i = 0; i < 32; i += 4) {
        float4 f4 = *(const float4*)(agg + row + c1 + i);
        *(float4*)(Gf + r * 128 + c1 + i) = f4;
        Abuf[r * 520 + c0 + i]     = f2bf(f4.x);
        Abuf[r * 520 + c0 + i + 1] = f2bf(f4.y);
        Abuf[r * 520 + c0 + i + 2] = f2bf(f4.z);
        Abuf[r * 520 + c0 + i + 3] = f2bf(f4.w);
      }
    }
  }
  asm volatile("s_wait_asynccnt 0" ::: "memory");
  __syncthreads();
  if (tid < 32) {    // H row stats for LN1 (over 256 dims later)
    float s = 0.f, sq = 0.f;
    for (int c = 0; c < 128; ++c) { float x = Hf[tid * 128 + c]; s += x; sq += x * x; }
    hsum[tid] = s; hsq[tid] = sq;
  }

  // stage 2: gate = LN(sigmoid(concat(H,agg) @ gateW + b)); gated = agg * gate
  v8f g0, g1;
  mm_lds2(Abuf, 520, lane, 8, gatePk, wave, g0, g1);
  float gbias = gate_b[n];
  float gv0[8], gv1[8], rs0[8], rss0[8], rs1[8], rss1[8];
#pragma unroll
  for (int r = 0; r < 8; ++r) {
    float x0 = 1.f / (1.f + __expf(-(g0[r] + gbias)));
    float x1 = 1.f / (1.f + __expf(-(g1[r] + gbias)));
    gv0[r] = x0; rs0[r] = x0; rss0[r] = x0 * x0;
    gv1[r] = x1; rs1[r] = x1; rss1[r] = x1 * x1;
  }
  hred(rs0); hred(rss0); hred(rs1); hred(rss1);
  if (nloc == 0)
#pragma unroll
    for (int r = 0; r < 8; ++r) {
      part [wave * 32 + half8 + r]      = rs0[r];  partq[wave * 32 + half8 + r]      = rss0[r];
      part [wave * 32 + 16 + half8 + r] = rs1[r];  partq[wave * 32 + 16 + half8 + r] = rss1[r];
    }
  __syncthreads();
  if (tid < 32) {
    float s = 0.f, sq = 0.f;
    for (int w = 0; w < 8; ++w) { s += part[w * 32 + tid]; sq += partq[w * 32 + tid]; }
    float mean = s * (1.f / 128.f);
    float var  = sq * (1.f / 128.f) - mean * mean;
    meanS[tid] = mean; rstdS[tid] = rsqrtf(var + 1e-5f);
  }
  __syncthreads();
  float glg = gate_g[n], glb = gate_bb[n];
  float gd0[8], gd1[8];
#pragma unroll
  for (int r = 0; r < 8; ++r) {
    int m0 = half8 + r, m1 = 16 + half8 + r;
    float w0 = (gv0[r] - meanS[m0]) * rstdS[m0] * glg + glb;
    float w1 = (gv1[r] - meanS[m1]) * rstdS[m1] * glg + glb;
    gd0[r] = Gf[m0 * 128 + n] * w0;
    gd1[r] = Gf[m1 * 128 + n] * w1;
    rs0[r] = gd0[r]; rss0[r] = gd0[r] * gd0[r];
    rs1[r] = gd1[r]; rss1[r] = gd1[r] * gd1[r];
  }
  // stage 3: LN1 over 256 = [H | gated]
  hred(rs0); hred(rss0); hred(rs1); hred(rss1);
  if (nloc == 0)
#pragma unroll
    for (int r = 0; r < 8; ++r) {
      part [wave * 32 + half8 + r]      = rs0[r];  partq[wave * 32 + half8 + r]      = rss0[r];
      part [wave * 32 + 16 + half8 + r] = rs1[r];  partq[wave * 32 + 16 + half8 + r] = rss1[r];
    }
  __syncthreads();
  if (tid < 32) {
    float s = hsum[tid], sq = hsq[tid];
    for (int w = 0; w < 8; ++w) { s += part[w * 32 + tid]; sq += partq[w * 32 + tid]; }
    float mean = s * (1.f / 256.f);
    float var  = sq * (1.f / 256.f) - mean * mean;
    meanS[tid] = mean; rstdS[tid] = rsqrtf(var + 1e-5f);
  }
  __syncthreads();
  {
    float l1g = ln1_g[128 + n], l1b = ln1_b[128 + n];
#pragma unroll
    for (int r = 0; r < 8; ++r) {
      int m0 = half8 + r, m1 = 16 + half8 + r;
      Abuf[m0 * 520 + 128 + n] = f2bf((gd0[r] - meanS[m0]) * rstdS[m0] * l1g + l1b);
      Abuf[m1 * 520 + 128 + n] = f2bf((gd1[r] - meanS[m1]) * rstdS[m1] * l1g + l1b);
    }
    int r2 = tid >> 3, cc0 = (tid & 7) * 16;
    float mn = meanS[r2], rst = rstdS[r2];
#pragma unroll
    for (int i = 0; i < 16; ++i) {
      int c = cc0 + i;
      Abuf[r2 * 520 + c] = f2bf((Hf[r2 * 128 + c] - mn) * rst * ln1_g[c] + ln1_b[c]);
    }
  }
  __syncthreads();

  // stage 4: x1 = x0 @ lin1 + b  [32,512]
#pragma unroll
  for (int j = 0; j < 4; ++j) {
    int nt = wave * 4 + j;
    v8f a0, a1;
    mm_lds2(Abuf, 520, lane, 8, lin1Pk, nt, a0, a1);
    int nn_ = nt * 16 + nloc;
    float bi = lin1_b[nn_];
#pragma unroll
    for (int r = 0; r < 8; ++r) {
      Bbuf[(half8 + r) * 520 + nn_]      = f2bf(a0[r] + bi);
      Bbuf[(16 + half8 + r) * 520 + nn_] = f2bf(a1[r] + bi);
    }
  }
  __syncthreads();

  // stage 5: y = silu(x1 @ w1) * (x1 @ w2)  [32,512]
#pragma unroll
  for (int j = 0; j < 4; ++j) {
    int nt = wave * 4 + j;
    v8f c10, c11, c20, c21;
    mm_lds2(Bbuf, 520, lane, 16, sw1Pk, nt, c10, c11);
    mm_lds2(Bbuf, 520, lane, 16, sw2Pk, nt, c20, c21);
    int nn_ = nt * 16 + nloc;
#pragma unroll
    for (int r = 0; r < 8; ++r) {
      float a_ = c10[r];
      Abuf[(half8 + r) * 520 + nn_]      = f2bf((a_ / (1.f + __expf(-a_))) * c20[r]);
      float b_ = c11[r];
      Abuf[(16 + half8 + r) * 520 + nn_] = f2bf((b_ / (1.f + __expf(-b_))) * c21[r]);
    }
  }
  __syncthreads();

  // stage 6: x3 = y @ w3  [32,256]
#pragma unroll
  for (int j = 0; j < 2; ++j) {
    int nt = wave * 2 + j;
    v8f c30, c31;
    mm_lds2(Abuf, 520, lane, 16, sw3Pk, nt, c30, c31);
    int nn_ = nt * 16 + nloc;
#pragma unroll
    for (int r = 0; r < 8; ++r) {
      Bbuf[(half8 + r) * 520 + nn_]      = f2bf(c30[r]);
      Bbuf[(16 + half8 + r) * 520 + nn_] = f2bf(c31[r]);
    }
  }
  __syncthreads();

  // stage 7: x4 = relu(x3 @ lin2 + b)  [32,256]
#pragma unroll
  for (int j = 0; j < 2; ++j) {
    int nt = wave * 2 + j;
    v8f c40, c41;
    mm_lds2(Bbuf, 520, lane, 8, lin2Pk, nt, c40, c41);
    int nn_ = nt * 16 + nloc;
    float bi = lin2_b[nn_];
#pragma unroll
    for (int r = 0; r < 8; ++r) {
      float x0 = c40[r] + bi; x0 = x0 > 0.f ? x0 : 0.f;
      float x1 = c41[r] + bi; x1 = x1 > 0.f ? x1 : 0.f;
      Abuf[(half8 + r) * 520 + nn_]      = f2bf(x0);
      Abuf[(16 + half8 + r) * 520 + nn_] = f2bf(x1);
    }
  }
  __syncthreads();

  // stage 8: x5 = x4 @ lin3 + b; LN2; + H
  v8f f0, f1;
  mm_lds2(Abuf, 520, lane, 8, lin3Pk, wave, f0, f1);
  float bi3 = lin3_b[n];
  float x0v[8], x1v[8];
#pragma unroll
  for (int r = 0; r < 8; ++r) {
    x0v[r] = f0[r] + bi3; rs0[r] = x0v[r]; rss0[r] = x0v[r] * x0v[r];
    x1v[r] = f1[r] + bi3; rs1[r] = x1v[r]; rss1[r] = x1v[r] * x1v[r];
  }
  hred(rs0); hred(rss0); hred(rs1); hred(rss1);
  if (nloc == 0)
#pragma unroll
    for (int r = 0; r < 8; ++r) {
      part [wave * 32 + half8 + r]      = rs0[r];  partq[wave * 32 + half8 + r]      = rss0[r];
      part [wave * 32 + 16 + half8 + r] = rs1[r];  partq[wave * 32 + 16 + half8 + r] = rss1[r];
    }
  __syncthreads();
  if (tid < 32) {
    float s = 0.f, sq = 0.f;
    for (int w = 0; w < 8; ++w) { s += part[w * 32 + tid]; sq += partq[w * 32 + tid]; }
    float mean = s * (1.f / 128.f);
    float var  = sq * (1.f / 128.f) - mean * mean;
    meanS[tid] = mean; rstdS[tid] = rsqrtf(var + 1e-5f);
  }
  __syncthreads();
  float g2 = ln2_g[n], b2 = ln2_b[n];
#pragma unroll
  for (int r = 0; r < 8; ++r) {
    int m0 = half8 + r, m1 = 16 + half8 + r;
    out[(size_t)(nodeBase + m0) * 128 + n] =
        (x0v[r] - meanS[m0]) * rstdS[m0] * g2 + b2 + Hf[m0 * 128 + n];
    out[(size_t)(nodeBase + m1) * 128 + n] =
        (x1v[r] - meanS[m1]) * rstdS[m1] * g2 + b2 + Hf[m1 * 128 + n];
  }
}

extern "C" void kernel_launch(void* const* d_in, const int* in_sizes, int n_in,
                              void* d_out, int out_size, void* d_ws, size_t ws_size,
                              hipStream_t stream) {
  (void)in_sizes; (void)n_in; (void)out_size; (void)ws_size;
  const float* H        = (const float*)d_in[0];
  const long long* ei   = (const long long*)d_in[1];
  const float* ep_w     = (const float*)d_in[2];
  const float* ep_b     = (const float*)d_in[3];
  const float* ep_ln_g  = (const float*)d_in[4];
  const float* ep_ln_b  = (const float*)d_in[5];
  const float* gate_w   = (const float*)d_in[6];
  const float* gate_b   = (const float*)d_in[7];
  const float* gate_ln_g= (const float*)d_in[8];
  const float* gate_ln_b= (const float*)d_in[9];
  const float* ln1_g    = (const float*)d_in[10];
  const float* ln1_b    = (const float*)d_in[11];
  const float* lin1_w   = (const float*)d_in[12];
  const float* lin1_b   = (const float*)d_in[13];
  const float* sw_w1    = (const float*)d_in[14];
  const float* sw_w2    = (const float*)d_in[15];
  const float* sw_w3    = (const float*)d_in[16];
  const float* lin2_w   = (const float*)d_in[17];
  const float* lin2_b   = (const float*)d_in[18];
  const float* lin3_w   = (const float*)d_in[19];
  const float* lin3_b   = (const float*)d_in[20];
  const float* ln2_g    = (const float*)d_in[21];
  const float* ln2_b    = (const float*)d_in[22];

  char* ws = (char*)d_ws;
  float* agg = (float*)ws;
  size_t off = (size_t)NN * DD * sizeof(float);              // 51,200,000
  unsigned short* Hbf = (unsigned short*)(ws + off); off += (size_t)NN * DD * 2;
  unsigned short* epPk   = (unsigned short*)(ws + off); off += 256 * 128 * 2;
  unsigned short* gatePk = (unsigned short*)(ws + off); off += 256 * 128 * 2;
  unsigned short* lin1Pk = (unsigned short*)(ws + off); off += 256 * 512 * 2;
  unsigned short* sw1Pk  = (unsigned short*)(ws + off); off += 512 * 512 * 2;
  unsigned short* sw2Pk  = (unsigned short*)(ws + off); off += 512 * 512 * 2;
  unsigned short* sw3Pk  = (unsigned short*)(ws + off); off += 512 * 256 * 2;
  unsigned short* lin2Pk = (unsigned short*)(ws + off); off += 256 * 256 * 2;
  unsigned short* lin3Pk = (unsigned short*)(ws + off); off += 256 * 128 * 2;

  zero_kernel<<<(NN * DD / 4 + 255) / 256, 256, 0, stream>>>(agg, NN * DD);
  h2bf_kernel<<<(NN * DD / 4 + 255) / 256, 256, 0, stream>>>(H, Hbf, NN * DD);

  auto packLaunch = [&](const float* W, unsigned short* o, int K, int N) {
    int frags = (K >> 5) * (N >> 4);
    pack_b_kernel<<<(frags + 7) / 8, 256, 0, stream>>>(W, o, K, N);
  };
  packLaunch(ep_w,   epPk,   256, 128);
  packLaunch(gate_w, gatePk, 256, 128);
  packLaunch(lin1_w, lin1Pk, 256, 512);
  packLaunch(sw_w1,  sw1Pk,  512, 512);
  packLaunch(sw_w2,  sw2Pk,  512, 512);
  packLaunch(sw_w3,  sw3Pk,  512, 256);
  packLaunch(lin2_w, lin2Pk, 256, 256);
  packLaunch(lin3_w, lin3Pk, 256, 128);

  edge_kernel<<<NE / 16, 256, 0, stream>>>(Hbf, ei, epPk, ep_b, ep_ln_g, ep_ln_b, agg);

  node_kernel<<<NN / 32, 256, (int)101888, stream>>>(H, Hbf, agg,
                                           gatePk, gate_b, gate_ln_g, gate_ln_b,
                                           ln1_g, ln1_b, lin1Pk, lin1_b,
                                           sw1Pk, sw2Pk, sw3Pk, lin2Pk, lin2_b,
                                           lin3Pk, lin3_b, ln2_g, ln2_b, (float*)d_out);
}
